// MultiHeadedSelfAttention_77025943486652
// MI455X (gfx1250) — matrix operand order repaired and checked by reading, not compiled, not run on hardware
//
#include <hip/hip_runtime.h>
#include <hip/hip_bf16.h>
#include <math.h>

#define HIDDEN 1024
#define NHEAD  16
#define DHEAD  64
#define SEQ    2048
#define BATCH  2
#define OUTD   1024

typedef __bf16 bf16_t;
typedef __attribute__((ext_vector_type(2)))  __bf16 bf16x2;
typedef __attribute__((ext_vector_type(16))) __bf16 v16bf;
typedef __attribute__((ext_vector_type(8)))  float  v8f;

__device__ __forceinline__ __bf16 to_bf16(float f) {
  unsigned u = __builtin_bit_cast(unsigned, f);
  u += 0x7FFFu + ((u >> 16) & 1u);            // round-to-nearest-even
  unsigned short h = (unsigned short)(u >> 16);
  return __builtin_bit_cast(__bf16, h);
}

__device__ __forceinline__ v8f wmma_bf16(v16bf a, v16bf b, v8f c) {
  return __builtin_amdgcn_wmma_f32_16x16x32_bf16(false, a, false, b, (short)0, c,
                                                 false, false);
}

// A fragment, M=16 x K=32 bf16, src row-major [m][k] with leading dim ldm.
// ISA layout: lane = 16*half + (m=l%16); VGPR j<4: K = 8*half+2j,+1 ;
// VGPR j>=4: K = 16 + 8*half + 2(j-4),+1
__device__ __forceinline__ v16bf load_frag_A(const bf16_t* __restrict__ base, int ldm) {
  const int lane = threadIdx.x & 31;
  const int m = lane & 15, half = lane >> 4;
  const bf16_t* row = base + (size_t)m * ldm;
  v16bf f;
#pragma unroll
  for (int j = 0; j < 4; ++j) {
    bf16x2 p = *reinterpret_cast<const bf16x2*>(row + half * 8 + 2 * j);
    f[2 * j] = p[0]; f[2 * j + 1] = p[1];
  }
#pragma unroll
  for (int j = 0; j < 4; ++j) {
    bf16x2 p = *reinterpret_cast<const bf16x2*>(row + 16 + half * 8 + 2 * j);
    f[8 + 2 * j] = p[0]; f[8 + 2 * j + 1] = p[1];
  }
  return f;
}

// B fragment, K=32 x N=16 bf16, where B[k][n] = src[n][k], src row-major [n][k]
// with leading dim ldn. ISA layout: lane n=l%16, half=l>>4; VGPR j<4: K=8*half+2j,+1 ; VGPR j>=4: K=16+8*half+2(j-4),+1
__device__ __forceinline__ v16bf load_frag_B(const bf16_t* __restrict__ base, int ldn) {
  const int lane = threadIdx.x & 31;
  const int n = lane & 15, half = lane >> 4;
  const bf16_t* row = base + (size_t)n * ldn;
  v16bf f;
#pragma unroll
  for (int j = 0; j < 8; ++j) {
    bf16x2 p = *reinterpret_cast<const bf16x2*>(row + half * 8 + 2 * j + (j >= 4 ? 8 : 0));
    f[2 * j] = p[0]; f[2 * j + 1] = p[1];
  }
  return f;
}

// ---------------- conversion kernels ----------------
__global__ __launch_bounds__(256) void k_cvt(const float* __restrict__ src,
                                             bf16_t* __restrict__ dst, int n) {
  int stride = gridDim.x * blockDim.x;
  for (int i = blockIdx.x * blockDim.x + threadIdx.x; i < n; i += stride)
    dst[i] = to_bf16(src[i]);
}

// src [rows][cols] f32 -> dst [cols][rows] bf16   (W [K][N] -> W^T [N][K])
__global__ __launch_bounds__(256) void k_cvt_t(const float* __restrict__ src,
                                               bf16_t* __restrict__ dst,
                                               int rows, int cols) {
  int n = rows * cols;
  int stride = gridDim.x * blockDim.x;
  for (int i = blockIdx.x * blockDim.x + threadIdx.x; i < n; i += stride) {
    int r = i / cols, c = i % cols;
    dst[(size_t)c * rows + r] = to_bf16(src[i]);
  }
}

// ---------------- software-pipelined 32Mx64N GEMM core ----------------
// acc[mi][nf] += A(32x1024) x B; A row-major [m][k], B^T row-major [n][k].
__device__ __forceinline__ void gemm_32x64(const bf16_t* __restrict__ Abase,
                                           const bf16_t* __restrict__ Bbase,
                                           v8f acc[2][4]) {
  v16bf a0 = load_frag_A(Abase, HIDDEN);
  v16bf a1 = load_frag_A(Abase + 16 * HIDDEN, HIDDEN);
  v16bf b0 = load_frag_B(Bbase + 0 * 16 * HIDDEN, HIDDEN);
  v16bf b1 = load_frag_B(Bbase + 1 * 16 * HIDDEN, HIDDEN);
  v16bf b2 = load_frag_B(Bbase + 2 * 16 * HIDDEN, HIDDEN);
  v16bf b3 = load_frag_B(Bbase + 3 * 16 * HIDDEN, HIDDEN);
  for (int k0 = 32; k0 < HIDDEN; k0 += 32) {
    // issue next-iteration loads before consuming current fragments
    v16bf na0 = load_frag_A(Abase + k0, HIDDEN);
    v16bf na1 = load_frag_A(Abase + 16 * HIDDEN + k0, HIDDEN);
    v16bf nb0 = load_frag_B(Bbase + 0 * 16 * HIDDEN + k0, HIDDEN);
    v16bf nb1 = load_frag_B(Bbase + 1 * 16 * HIDDEN + k0, HIDDEN);
    v16bf nb2 = load_frag_B(Bbase + 2 * 16 * HIDDEN + k0, HIDDEN);
    v16bf nb3 = load_frag_B(Bbase + 3 * 16 * HIDDEN + k0, HIDDEN);
    acc[0][0] = wmma_bf16(a0, b0, acc[0][0]);
    acc[0][1] = wmma_bf16(a0, b1, acc[0][1]);
    acc[0][2] = wmma_bf16(a0, b2, acc[0][2]);
    acc[0][3] = wmma_bf16(a0, b3, acc[0][3]);
    acc[1][0] = wmma_bf16(a1, b0, acc[1][0]);
    acc[1][1] = wmma_bf16(a1, b1, acc[1][1]);
    acc[1][2] = wmma_bf16(a1, b2, acc[1][2]);
    acc[1][3] = wmma_bf16(a1, b3, acc[1][3]);
    a0 = na0; a1 = na1; b0 = nb0; b1 = nb1; b2 = nb2; b3 = nb3;
  }
  acc[0][0] = wmma_bf16(a0, b0, acc[0][0]);
  acc[0][1] = wmma_bf16(a0, b1, acc[0][1]);
  acc[0][2] = wmma_bf16(a0, b2, acc[0][2]);
  acc[0][3] = wmma_bf16(a0, b3, acc[0][3]);
  acc[1][0] = wmma_bf16(a1, b0, acc[1][0]);
  acc[1][1] = wmma_bf16(a1, b1, acc[1][1]);
  acc[1][2] = wmma_bf16(a1, b2, acc[1][2]);
  acc[1][3] = wmma_bf16(a1, b3, acc[1][3]);
}

// ---------------- QKV projection GEMM ----------------
// vtrans==0 -> dst = q/k laid out [B,H,S,D]; vtrans==1 -> dst = vT [B,H,D,S]
__global__ __launch_bounds__(256) void k_proj_qkv(const bf16_t* __restrict__ xb,
                                                  const bf16_t* __restrict__ wT,
                                                  const float*  __restrict__ bias,
                                                  bf16_t* __restrict__ dst,
                                                  int vtrans) {
  const int lane = threadIdx.x & 31;
  const int warp = threadIdx.x >> 5;
  const int wid  = blockIdx.x * 8 + warp;   // 0..2047
  const int m0   = (wid >> 4) * 32;         // 32-row tile
  const int n0   = (wid & 15) * 64;         // 64-wide col chunk

  v8f acc[2][4] = {};
  gemm_32x64(xb + (size_t)m0 * HIDDEN, wT + (size_t)n0 * HIDDEN, acc);

  const int half = lane >> 4, nl = lane & 15;
#pragma unroll
  for (int nf = 0; nf < 4; ++nf) {
    const int col = n0 + nf * 16 + nl;
    const float bia = bias[col];
    const int h = col >> 6, d = col & 63;
#pragma unroll
    for (int mi = 0; mi < 2; ++mi) {
#pragma unroll
      for (int r = 0; r < 8; ++r) {
        const int m = m0 + mi * 16 + r + 8 * half;
        const int bb = m >> 11, s = m & 2047;  // m = b*SEQ + s
        const float v = acc[mi][nf][r] + bia;
        size_t idx = vtrans
            ? ((size_t)((bb * NHEAD + h) * DHEAD + d)) * SEQ + s
            : ((size_t)((bb * NHEAD + h) * SEQ + s)) * DHEAD + d;
        dst[idx] = to_bf16(v);
      }
    }
  }
}

// ---------------- flash attention ----------------
// Per-chunk computation: 32 keys against one 16-query tile (online softmax).
__device__ __forceinline__ void attn_chunk(const v16bf ka[4], const v16bf vb[4],
                                           const v16bf& qb0, const v16bf& qb1,
                                           v8f o[4], float& m_run, float& l_run,
                                           int lane) {
  const int half = lane >> 4;
  const float scale = 0.125f;               // 1/sqrt(64)

  v8f s0 = {};
  s0 = wmma_bf16(ka[0], qb0, s0);
  s0 = wmma_bf16(ka[1], qb1, s0);
  v8f s1 = {};
  s1 = wmma_bf16(ka[2], qb0, s1);
  s1 = wmma_bf16(ka[3], qb1, s1);

  // lane holds 16 key-scores of query (lane%16); partner lane^16 has the rest
  float mloc = -INFINITY;
#pragma unroll
  for (int r = 0; r < 8; ++r) {
    s0[r] *= scale; s1[r] *= scale;
    mloc = fmaxf(mloc, fmaxf(s0[r], s1[r]));
  }
  mloc = fmaxf(mloc, __shfl_xor(mloc, 16, 32));
  const float m_new = fmaxf(m_run, mloc);
  const float corr  = __expf(m_run - m_new);

  float p0[8], p1[8], lloc = 0.f;
#pragma unroll
  for (int r = 0; r < 8; ++r) {
    p0[r] = __expf(s0[r] - m_new);
    p1[r] = __expf(s1[r] - m_new);
    lloc += p0[r] + p1[r];
  }
  lloc += __shfl_xor(lloc, 16, 32);
  l_run = l_run * corr + lloc;
  m_run = m_new;

  // rescale O rows (row m = r + 8*half is a QUERY; its corr lives in lane m)
#pragma unroll
  for (int r = 0; r < 8; ++r) {
    const float cr = __shfl(corr, r + 8 * half, 32);
#pragma unroll
    for (int nf = 0; nf < 4; ++nf) o[nf][r] *= cr;
  }

  // S^T C-layout maps bit-for-bit to P^T A-layout (query x 32key): pack exps
  v16bf pf;
#pragma unroll
  for (int j = 0; j < 4; ++j) {
    pf[2 * j]         = to_bf16(p0[2 * j]);
    pf[2 * j + 1]     = to_bf16(p0[2 * j + 1]);
    pf[8 + 2 * j]     = to_bf16(p1[2 * j]);
    pf[8 + 2 * j + 1] = to_bf16(p1[2 * j + 1]);
  }

#pragma unroll
  for (int nf = 0; nf < 4; ++nf) o[nf] = wmma_bf16(pf, vb[nf], o[nf]);
}

__device__ __forceinline__ void load_kv(const bf16_t* __restrict__ kbase,
                                        const bf16_t* __restrict__ vbase,
                                        int kc, v16bf ka[4], v16bf vb[4]) {
  ka[0] = load_frag_A(kbase + (size_t)kc * DHEAD + 0,  DHEAD);
  ka[1] = load_frag_A(kbase + (size_t)kc * DHEAD + 32, DHEAD);
  ka[2] = load_frag_A(kbase + (size_t)(kc + 16) * DHEAD + 0,  DHEAD);
  ka[3] = load_frag_A(kbase + (size_t)(kc + 16) * DHEAD + 32, DHEAD);
#pragma unroll
  for (int nf = 0; nf < 4; ++nf)
    vb[nf] = load_frag_B(vbase + (size_t)(nf * 16) * SEQ + kc, SEQ);
}

// One wave owns one 16-query tile of one (b,h).
__global__ __launch_bounds__(256) void k_attn(const bf16_t* __restrict__ q,
                                              const bf16_t* __restrict__ kk,
                                              const bf16_t* __restrict__ vT,
                                              bf16_t* __restrict__ concat) {
  const int lane = threadIdx.x & 31;
  const int warp = threadIdx.x >> 5;
  const int bh = blockIdx.y;                 // 0..31
  const int bb = bh >> 4, h = bh & 15;
  const int qt = blockIdx.x * 8 + warp;      // 0..127
  const int half = lane >> 4, nl = lane & 15;

  const bf16_t* qbase = q  + ((size_t)bh * SEQ + qt * 16) * DHEAD;
  const bf16_t* kbase = kk + (size_t)bh * SEQ * DHEAD;
  const bf16_t* vbase = vT + (size_t)bh * DHEAD * SEQ;

  const v16bf qb0 = load_frag_B(qbase + 0,  DHEAD);   // B[d][query], d=0..31
  const v16bf qb1 = load_frag_B(qbase + 32, DHEAD);   // d=32..63

  v8f o[4] = {};
  float m_run = -INFINITY, l_run = 0.f;

  v16bf ka[4], vb[4];
  load_kv(kbase, vbase, 0, ka, vb);
  for (int kc = 32; kc < SEQ; kc += 32) {
    v16bf nka[4], nvb[4];
    load_kv(kbase, vbase, kc, nka, nvb);     // overlap with current chunk's math
    attn_chunk(ka, vb, qb0, qb1, o, m_run, l_run, lane);
#pragma unroll
    for (int i = 0; i < 4; ++i) { ka[i] = nka[i]; vb[i] = nvb[i]; }
  }
  attn_chunk(ka, vb, qb0, qb1, o, m_run, l_run, lane);

  // epilogue: divide by row sum, write concat[b*S+s][h*64+dv] in bf16
#pragma unroll
  for (int r = 0; r < 8; ++r) {
    const float lr = __shfl(l_run, r + 8 * half, 32);
    const int s = qt * 16 + r + 8 * half;
    const size_t rowbase = ((size_t)(bb * SEQ + s)) * (NHEAD * DHEAD) + h * DHEAD;
#pragma unroll
    for (int nf = 0; nf < 4; ++nf)
      concat[rowbase + nf * 16 + nl] = to_bf16(o[nf][r] / lr);
  }
}

// ---------------- output projection ----------------
__global__ __launch_bounds__(256) void k_out(const bf16_t* __restrict__ cb,
                                             const bf16_t* __restrict__ woT,
                                             const float* __restrict__ bo,
                                             float* __restrict__ out) {
  const int lane = threadIdx.x & 31;
  const int warp = threadIdx.x >> 5;
  const int wid  = blockIdx.x * 8 + warp;   // 0..2047
  const int m0   = (wid >> 4) * 32;
  const int n0   = (wid & 15) * 64;

  v8f acc[2][4] = {};
  gemm_32x64(cb + (size_t)m0 * HIDDEN, woT + (size_t)n0 * HIDDEN, acc);

  const int half = lane >> 4, nl = lane & 15;
#pragma unroll
  for (int nf = 0; nf < 4; ++nf) {
    const int col = n0 + nf * 16 + nl;
    const float bia = bo[col];
#pragma unroll
    for (int mi = 0; mi < 2; ++mi) {
#pragma unroll
      for (int r = 0; r < 8; ++r) {
        const int m = m0 + mi * 16 + r + 8 * half;
        out[(size_t)m * OUTD + col] = acc[mi][nf][r] + bia;
      }
    }
  }
}

extern "C" void kernel_launch(void* const* d_in, const int* in_sizes, int n_in,
                              void* d_out, int out_size, void* d_ws, size_t ws_size,
                              hipStream_t stream) {
  const float* x  = (const float*)d_in[0];
  const float* Wq = (const float*)d_in[1];
  const float* bq = (const float*)d_in[2];
  const float* Wk = (const float*)d_in[3];
  const float* bk = (const float*)d_in[4];
  const float* Wv = (const float*)d_in[5];
  const float* bv = (const float*)d_in[6];
  const float* Wo = (const float*)d_in[7];
  const float* bo = (const float*)d_in[8];
  float* out = (float*)d_out;

  // workspace layout (bytes)
  char* ws = (char*)d_ws;
  const size_t OFF_XB  = 0;                       // 4096x1024 bf16 : 8 MiB
  const size_t OFF_WQT = OFF_XB  + 8388608;       // 1024x1024 bf16 : 2 MiB
  const size_t OFF_WKT = OFF_WQT + 2097152;
  const size_t OFF_WVT = OFF_WKT + 2097152;
  const size_t OFF_WOT = OFF_WVT + 2097152;
  const size_t OFF_Q   = OFF_WOT + 2097152;       // [B,H,S,D]  : 8 MiB
  const size_t OFF_K   = OFF_Q   + 8388608;       // [B,H,S,D]  : 8 MiB
  const size_t OFF_VT  = OFF_K   + 8388608;       // [B,H,D,S]  : 8 MiB
  const size_t OFF_CC  = OFF_VT  + 8388608;       // [4096,1024]: 8 MiB
  const size_t NEED    = OFF_CC  + 8388608;       // 48 MiB total
  if (ws_size < NEED) return;

  bf16_t* XB  = (bf16_t*)(ws + OFF_XB);
  bf16_t* WQT = (bf16_t*)(ws + OFF_WQT);
  bf16_t* WKT = (bf16_t*)(ws + OFF_WKT);
  bf16_t* WVT = (bf16_t*)(ws + OFF_WVT);
  bf16_t* WOT = (bf16_t*)(ws + OFF_WOT);
  bf16_t* Qb  = (bf16_t*)(ws + OFF_Q);
  bf16_t* Kb  = (bf16_t*)(ws + OFF_K);
  bf16_t* VTb = (bf16_t*)(ws + OFF_VT);
  bf16_t* CC  = (bf16_t*)(ws + OFF_CC);

  k_cvt  <<<1024, 256, 0, stream>>>(x,  XB, BATCH * SEQ * HIDDEN);
  k_cvt_t<<<512,  256, 0, stream>>>(Wq, WQT, HIDDEN, NHEAD * DHEAD);
  k_cvt_t<<<512,  256, 0, stream>>>(Wk, WKT, HIDDEN, NHEAD * DHEAD);
  k_cvt_t<<<512,  256, 0, stream>>>(Wv, WVT, HIDDEN, NHEAD * DHEAD);
  k_cvt_t<<<512,  256, 0, stream>>>(Wo, WOT, NHEAD * DHEAD, OUTD);

  k_proj_qkv<<<256, 256, 0, stream>>>(XB, WQT, bq, Qb, 0);
  k_proj_qkv<<<256, 256, 0, stream>>>(XB, WKT, bk, Kb, 0);
  k_proj_qkv<<<256, 256, 0, stream>>>(XB, WVT, bv, VTb, 1);

  dim3 agrid(16, 32);   // 16 blocks x 8 waves = 128 query tiles; 32 (b,h)
  k_attn<<<agrid, 256, 0, stream>>>(Qb, Kb, VTb, CC);

  k_out<<<256, 256, 0, stream>>>(CC, WOT, bo, out);
}
